// XAttn_68659347194155
// MI455X (gfx1250) — matrix operand order repaired and checked by reading, not compiled, not run on hardware
//
#include <hip/hip_runtime.h>
#include <hip/hip_bf16.h>
#include <stdint.h>

typedef __attribute__((ext_vector_type(16))) __bf16 v16bf;
typedef __attribute__((ext_vector_type(8)))  float  v8f;

constexpr int kHeads = 8;
constexpr int kDin   = 128;
constexpr int kDkey  = 32;
constexpr float kEps = 1e-5f;
constexpr float kInvTemp = 0.17677669529663688f; // 1/sqrt(32)

// fragment sizes in uint32 words (each fragment-lane: 16 bf16 = 8 words)
constexpr size_t kWFragWords = (size_t)64 * 4 * 32 * 8;      // 65536 words = 256KB per weight matrix
constexpr size_t kCFragWords = (size_t)kHeads * 2 * 4 * 32 * 8; // 16384 words = 64KB per c matrix

union VB { v16bf v; uint32_t u[8]; };
union VF { v8f  v; float    f[8]; };

__device__ __forceinline__ uint32_t pack_bf16(float lo, float hi) {
  uint32_t a = __float_as_uint(lo);
  uint32_t b = __float_as_uint(hi);
  a = (a + 0x7FFFu + ((a >> 16) & 1u)) >> 16;   // RNE f32 -> bf16
  b = (b + 0x7FFFu + ((b >> 16) & 1u)) >> 16;
  return (a & 0xFFFFu) | (b << 16);
}

struct SMem {
  uint32_t a_h[16 * 64];            // bf16 [16][128] A tile (h_emb)
  uint32_t a_r[16 * 64];            // bf16 [16][128] A tile (r_emb)
  float hk[16 * kHeads * kDkey];    // h1k fp32
  float rk[16 * kHeads * kDkey];    // r1k fp32
  float tbuf[16 * kDin];            // head-combined accumulator
  float xbuf[16 * 64];              // softmax numerators
  float red[16 * 16];
  float red2[16 * 16];
  float rowA[16];
  float rowB[16];
  float hx[16 * kHeads];
  float rx[16 * kHeads];
};

// A fragment (16x32 bf16, MxK) from LDS packed-bf16 [16][128]
__device__ __forceinline__ v16bf load_A(const uint32_t* a, int lane, int ks) {
  const int m  = lane & 15;
  const int k0 = ks * 32 + ((lane >> 4) ? 16 : 0);     // element index
  const uint32_t* p = a + m * 64 + (k0 >> 1);          // packed-pair index
  VB t;
  t.u[0] = p[0]; t.u[1] = p[1]; t.u[2] = p[2]; t.u[3] = p[3];   // K .. K+7
  t.u[4] = p[4]; t.u[5] = p[5]; t.u[6] = p[6]; t.u[7] = p[7]; // K+8 .. K+15
  return t.v;
}

// B fragment from pre-swizzled bf16 workspace: two coalesced b128 loads
__device__ __forceinline__ v16bf load_Bp(const uint4* base, int frag, int lane) {
  const uint4* p = base + ((size_t)frag * 32 + lane) * 2;
  uint4 a = p[0], b = p[1];
  VB t;
  t.u[0] = a.x; t.u[1] = a.y; t.u[2] = a.z; t.u[3] = a.w;
  t.u[4] = b.x; t.u[5] = b.y; t.u[6] = b.z; t.u[7] = b.w;
  return t.v;
}

__device__ __forceinline__ v8f wmma_bf16(v16bf a, v16bf b, v8f c) {
  return __builtin_amdgcn_wmma_f32_16x16x32_bf16(
      /*neg_a=*/false, a, /*neg_b=*/false, b,
      /*c_mod=*/(short)0, c, /*reuse_a=*/false, /*reuse_b=*/false);
}

// --------- prep 1: swizzle+convert weight matrices into B-fragment layout
// fragment index = T*4 + ks (T = 16-row tile of W); per lane: 16 bf16
__global__ __launch_bounds__(128) void xattn_pack_w(
    const float* __restrict__ w_h, const float* __restrict__ w_r,
    uint32_t* __restrict__ wp_h, uint32_t* __restrict__ wp_r) {
  const float* w    = blockIdx.y ? w_r : w_h;
  uint32_t* wp      = blockIdx.y ? wp_r : wp_h;
  const int T    = blockIdx.x;          // 0..63
  const int ks   = threadIdx.x >> 5;    // 0..3
  const int lane = threadIdx.x & 31;
  const int n    = lane & 15;
  const int kk   = ks * 32 + ((lane >> 4) ? 16 : 0);
  const float* p = w + (size_t)(T * 16 + n) * kDin + kk;
  uint32_t* q    = wp + ((size_t)(T * 4 + ks) * 32 + lane) * 8;
#pragma unroll
  for (int i = 0; i < 8; ++i) q[i] = pack_bf16(p[2 * i], p[2 * i + 1]);
}

// --------- prep 2: c[h] = k @ w_head, then swizzle+convert to B fragments
__global__ __launch_bounds__(256) void xattn_prep_c(
    const float* __restrict__ k_h, const float* __restrict__ w_h,
    const float* __restrict__ k_r, const float* __restrict__ w_r,
    uint32_t* __restrict__ cp_h, uint32_t* __restrict__ cp_r) {
  __shared__ float c[kDkey * kDin];
  const int head = blockIdx.x;
  const float* k = blockIdx.y ? k_r : k_h;
  const float* w = (blockIdx.y ? w_r : w_h) + (size_t)head * kDin * kDin;
  uint32_t* cp   = (blockIdx.y ? cp_r : cp_h) + (size_t)head * (2 * 4 * 32 * 8);

  for (int o = threadIdx.x; o < kDkey * kDin; o += 256) {
    int i = o >> 7, j = o & 127;
    float s = 0.f;
    for (int d = 0; d < kDin; ++d) s += k[i * kDin + d] * w[d * kDin + j];
    c[o] = s;
  }
  __syncthreads();
  // 2 n-tiles * 4 ks * 32 lanes = 256 fragment-lanes; one per thread
  const int t    = threadIdx.x;
  const int lane = t & 31;
  const int ks   = (t >> 5) & 3;
  const int nt   = t >> 7;
  const int n    = lane & 15;
  const int kk   = ks * 32 + ((lane >> 4) ? 16 : 0);
  const float* p = &c[(nt * 16 + n) * kDin + kk];
  uint32_t* q    = cp + (size_t)t * 8;
#pragma unroll
  for (int i = 0; i < 8; ++i) q[i] = pack_bf16(p[2 * i], p[2 * i + 1]);
}

// --------- main fused kernel: one 16-row batch tile per workgroup
__global__ __launch_bounds__(256) void xattn_main(
    const float* __restrict__ h_emb, const float* __restrict__ r_emb,
    const uint4* __restrict__ wp_h, const uint4* __restrict__ wp_r,
    const uint4* __restrict__ cp_h, const uint4* __restrict__ cp_r,
    const float* __restrict__ gamma, const float* __restrict__ beta,
    float* __restrict__ out) {
  __shared__ SMem sm;
  const int tid  = threadIdx.x;
  const int wave = tid >> 5;     // head index
  const int lane = tid & 31;
  const long g0  = (long)blockIdx.x * 16;

  // ---- stage input tiles as packed bf16; zero combine buffer
  for (int idx = tid; idx < 1024; idx += 256) {
    int rr = idx >> 6, cp = idx & 63;
    const float* ph = h_emb + (g0 + rr) * kDin + 2 * cp;
    const float* pr = r_emb + (g0 + rr) * kDin + 2 * cp;
    sm.a_h[idx] = pack_bf16(ph[0], ph[1]);
    sm.a_r[idx] = pack_bf16(pr[0], pr[1]);
  }
  for (int idx = tid; idx < 16 * kDin; idx += 256) sm.tbuf[idx] = 0.0f;
  __syncthreads();

  // ---- A fragments (reused across all N tiles)
  v16bf Ah[4], Ar[4];
#pragma unroll
  for (int ks = 0; ks < 4; ++ks) {
    Ah[ks] = load_A(sm.a_h, lane, ks);
    Ar[ks] = load_A(sm.a_r, lane, ks);
  }

  const v8f vzero = {0.f, 0.f, 0.f, 0.f, 0.f, 0.f, 0.f, 0.f};
  const int mh = (lane >> 4) * 8;   // VGPR r holds row M = mh + r of C tiles

  // ---- key projections: h1k/r1k = emb @ c[head].T  (16x32 per head)
#pragma unroll
  for (int nt = 0; nt < 2; ++nt) {
    v8f acc_hk = vzero, acc_rk = vzero;
#pragma unroll
    for (int ks = 0; ks < 4; ++ks) {
      const int frag = (wave * 2 + nt) * 4 + ks;
      acc_hk = wmma_bf16(Ah[ks], load_Bp(cp_h, frag, lane), acc_hk);
      acc_rk = wmma_bf16(Ar[ks], load_Bp(cp_r, frag, lane), acc_rk);
    }
    VF uh; uh.v = acc_hk;
    VF ur; ur.v = acc_rk;
    const int n = nt * 16 + (lane & 15);
#pragma unroll
    for (int r = 0; r < 8; ++r) {
      const int m = mh + r;
      sm.hk[(m * kHeads + wave) * kDkey + n] = uh.f[r];
      sm.rk[(m * kHeads + wave) * kDkey + n] = ur.f[r];
    }
  }

  // ---- head projections: wave i keeps h1/r1 head-i tiles in registers
  v8f acch[8], accr[8];
#pragma unroll
  for (int nt = 0; nt < 8; ++nt) {
    const int fb = (wave * 8 + nt) * 4;
    if (nt < 7) {  // pull next fragment block toward L0/L2 (global_prefetch_b8)
      __builtin_prefetch(wp_h + ((size_t)(fb + 4) * 32 + lane) * 2, 0, 0);
      __builtin_prefetch(wp_r + ((size_t)(fb + 4) * 32 + lane) * 2, 0, 0);
    }
    v8f ah = vzero, ar = vzero;
#pragma unroll
    for (int ks = 0; ks < 4; ++ks) {
      ah = wmma_bf16(Ah[ks], load_Bp(wp_h, fb + ks, lane), ah);
      ar = wmma_bf16(Ar[ks], load_Bp(wp_r, fb + ks, lane), ar);
    }
    acch[nt] = ah; accr[nt] = ar;
  }
  __syncthreads();

  // ---- attention: 16 threads per row, 4 of 64 logits each
  {
    const int row = tid >> 4, sub = tid & 15;
    float lg[4];
#pragma unroll
    for (int q = 0; q < 4; ++q) {
      const int f = sub * 4 + q;
      const float* ph = &sm.hk[(row * kHeads + (f >> 3)) * kDkey];
      const float* pr = &sm.rk[(row * kHeads + (f & 7)) * kDkey];
      float s = 0.f;
#pragma unroll
      for (int k = 0; k < kDkey; ++k) s += ph[k] * pr[k];
      lg[q] = s * kInvTemp;
    }
    float lm = fmaxf(fmaxf(lg[0], lg[1]), fmaxf(lg[2], lg[3]));
    sm.red[row * 16 + sub] = lm;
    __syncthreads();
    if (sub == 0) {
      float m = sm.red[row * 16];
      for (int i = 1; i < 16; ++i) m = fmaxf(m, sm.red[row * 16 + i]);
      sm.rowA[row] = m;
    }
    __syncthreads();
    const float rm = sm.rowA[row];
    float ps = 0.f;
#pragma unroll
    for (int q = 0; q < 4; ++q) {
      float e = __expf(lg[q] - rm);
      sm.xbuf[row * 64 + sub * 4 + q] = e;
      ps += e;
    }
    sm.red[row * 16 + sub] = ps;
    __syncthreads();
    if (sub == 0) {
      float s = 0.f;
      for (int i = 0; i < 16; ++i) s += sm.red[row * 16 + i];
      sm.rowB[row] = 1.0f / s;
    }
    __syncthreads();
    const float inv = sm.rowB[row];
    if (sub < 8) {
      float s = 0.f;
      for (int j = 0; j < 8; ++j) s += sm.xbuf[row * 64 + sub * 8 + j];
      sm.hx[row * 8 + sub] = s * inv;
    } else {
      const int j = sub - 8;
      float s = 0.f;
      for (int h = 0; h < 8; ++h) s += sm.xbuf[row * 64 + h * 8 + j];
      sm.rx[row * 8 + j] = s * inv;
    }
  }
  __syncthreads();

  // ---- weighted head combine: t += h1*hx + r1*rx (ds_add_f32 across heads)
  {
    float hxv[8], rxv[8];
#pragma unroll
    for (int r = 0; r < 8; ++r) {
      const int m = mh + r;
      hxv[r] = sm.hx[m * 8 + wave];
      rxv[r] = sm.rx[m * 8 + wave];
    }
#pragma unroll
    for (int nt = 0; nt < 8; ++nt) {
      VF uh; uh.v = acch[nt];
      VF ur; ur.v = accr[nt];
      const int n = nt * 16 + (lane & 15);
#pragma unroll
      for (int r = 0; r < 8; ++r) {
        const int m = mh + r;
        atomicAdd(&sm.tbuf[m * kDin + n], uh.f[r] * hxv[r] + ur.f[r] * rxv[r]);
      }
    }
  }
  __syncthreads();

  // ---- residual + LayerNorm + store
  {
    const int row = tid >> 4, sub = tid & 15;
    const float* res = h_emb + (g0 + row) * kDin;
    float v[8], s = 0.f, s2 = 0.f;
#pragma unroll
    for (int q = 0; q < 8; ++q) {
      const float x = sm.tbuf[row * kDin + sub * 8 + q] + res[sub * 8 + q];
      v[q] = x; s += x; s2 += x * x;
    }
    sm.red[row * 16 + sub]  = s;
    sm.red2[row * 16 + sub] = s2;
    __syncthreads();
    if (sub == 0) {
      float S = 0.f, S2 = 0.f;
      for (int i = 0; i < 16; ++i) { S += sm.red[row * 16 + i]; S2 += sm.red2[row * 16 + i]; }
      const float mu  = S * (1.0f / kDin);
      const float var = S2 * (1.0f / kDin) - mu * mu;
      sm.rowA[row] = mu;
      sm.rowB[row] = rsqrtf(var + kEps);
    }
    __syncthreads();
    const float mu = sm.rowA[row], is = sm.rowB[row];
    float* po = out + (g0 + row) * kDin;
#pragma unroll
    for (int q = 0; q < 8; ++q) {
      const int d = sub * 8 + q;
      po[d] = gamma[d] * (v[q] - mu) * is + beta[d];
    }
  }
}

extern "C" void kernel_launch(void* const* d_in, const int* in_sizes, int n_in,
                              void* d_out, int out_size, void* d_ws, size_t ws_size,
                              hipStream_t stream) {
  const float* h_emb = (const float*)d_in[0];
  const float* r_emb = (const float*)d_in[1];
  const float* w_h   = (const float*)d_in[2];
  const float* w_r   = (const float*)d_in[3];
  const float* k_h   = (const float*)d_in[4];
  const float* k_r   = (const float*)d_in[5];
  const float* gamma = (const float*)d_in[6];
  const float* beta  = (const float*)d_in[7];
  float* out = (float*)d_out;

  uint32_t* wp_h = (uint32_t*)d_ws;
  uint32_t* wp_r = wp_h + kWFragWords;
  uint32_t* cp_h = wp_r + kWFragWords;
  uint32_t* cp_r = cp_h + kCFragWords;

  const int nB = in_sizes[0] / kDin;                   // 65536

  xattn_pack_w<<<dim3(64, 2), 128, 0, stream>>>(w_h, w_r, wp_h, wp_r);
  xattn_prep_c<<<dim3(kHeads, 2), 256, 0, stream>>>(k_h, w_h, k_r, w_r, cp_h, cp_r);
  xattn_main<<<dim3(nB / 16), 256, 0, stream>>>(
      h_emb, r_emb, (const uint4*)wp_h, (const uint4*)wp_r,
      (const uint4*)cp_h, (const uint4*)cp_r, gamma, beta, out);
}
